// VPINN_46334107189603
// MI455X (gfx1250) — matrix operand order repaired and checked by reading, not compiled, not run on hardware
//
#include <hip/hip_runtime.h>

#define NPTS   32768
#define HDIM   256
#define KTEST  1024
#define NSPLIT 64
#define PI_F   3.14159265358979323846f

typedef __attribute__((ext_vector_type(16))) __bf16 v16bf;
typedef __attribute__((ext_vector_type(8)))  float  v8f;

__device__ __forceinline__ unsigned short f2bf(float f) {
    union { float f; unsigned u; } x; x.f = f;
    unsigned u = x.u;
    return (unsigned short)((u + 0x7FFFu + ((u >> 16) & 1u)) >> 16); // RNE
}
__device__ __forceinline__ float bf2f(unsigned short h) {
    union { unsigned u; float f; } x; x.u = ((unsigned)h) << 16;
    return x.f;
}

// Branch-free tanh: 2 transcendentals, no exec-mask divergence
// (libm tanhf lowers to a branchy saveexec region that stalls WMMA co-issue).
__device__ __forceinline__ float fast_tanh(float v) {
    float av = __builtin_fabsf(v);
    float e  = __expf(-2.0f * av);
    float t  = __fdividef(1.0f - e, 1.0f + e);
    return __builtin_copysignf(t, v);
}

// A-matrix fragment (16x32 bf16) per CDNA5 ISA layout:
// lane l (=lane&15) holds row M=l; half hi (=lane>>4) selects K chunks
// [kt*32+8*hi, +8) -> slots 0..7 and [kt*32+16+8*hi, +8) -> slots 8..15.
__device__ __forceinline__ v16bf load_afrag(const unsigned short* buf, int l, int hi, int kt) {
    union { v16bf v; unsigned short s[16]; } u;
    const unsigned short* p0 = buf + l * HDIM + kt * 32 + 8 * hi;
    const unsigned short* p1 = p0 + 16;
#pragma unroll
    for (int i = 0; i < 8; ++i) { u.s[i] = p0[i]; u.s[8 + i] = p1[i]; }
    return u.v;
}

// B-matrix fragment (32x16 bf16): lane L holds column N=nt*16+(L&15), K=kt*32+8*(L>>4)+(e&7)+16*(e>>3).
// Pre-packed contiguously: 32B per lane, 1KB per fragment.
__device__ __forceinline__ v16bf load_bfrag(const unsigned short* packW, int base, int kt, int nt, int lane) {
    union { v16bf v; unsigned short s[16]; } u;
    const unsigned short* p = packW + base + ((kt * 16 + nt) * 32 + lane) * 16;
#pragma unroll
    for (int i = 0; i < 16; ++i) u.s[i] = p[i];
    return u.v;
}

__global__ __launch_bounds__(32) void zero_out_kernel(float* o) {
    if (threadIdx.x < 2) o[threadIdx.x] = 0.0f;
}

// Re-layout W1, W2 (fin x fout row-major f32) into bf16 WMMA B-fragments.
// dst flat index t = [lay][kt][nt][lane][e].
__global__ __launch_bounds__(256) void pack_weights_kernel(
    const float* __restrict__ W1, const float* __restrict__ W2,
    unsigned short* __restrict__ dst)
{
    int t = blockIdx.x * blockDim.x + threadIdx.x;
    if (t >= 2 * 65536) return;
    int e    = t & 15;
    int lane = (t >> 4) & 31;
    int nt   = (t >> 9) & 15;
    int kt   = (t >> 13) & 7;
    int lay  = t >> 16;
    const float* W = lay ? W2 : W1;
    int K = kt * 32 + 8 * (lane >> 4) + (e & 7) + 16 * (e >> 3);
    int N = nt * 16 + (lane & 15);
    dst[t] = f2bf(W[K * HDIM + N]);
}

// Fused MLP + Taylor-mode forward-over-forward AD (value h, dh/dx, d2h/dx2).
// 2 waves / block, each wave owns a private 16-point tile end-to-end.
__global__ __launch_bounds__(64) void vpinn_mlp_kernel(
    const float* __restrict__ x,  const float* __restrict__ wq,
    const float* __restrict__ W0, const float* __restrict__ b0,
    const float* __restrict__ b1, const float* __restrict__ b2,
    const float* __restrict__ W3, const float* __restrict__ b3,
    const unsigned short* __restrict__ packW,
    float* __restrict__ vout, float* __restrict__ out)
{
    __shared__ unsigned short lds_a[2][16 * HDIM];
    __shared__ unsigned short lds_g[2][16 * HDIM];
    __shared__ unsigned short lds_s[2][16 * HDIM];

    const int wave = threadIdx.x >> 5;
    const int lane = threadIdx.x & 31;
    const int l  = lane & 15;
    const int hi = lane >> 4;
    unsigned short* A = lds_a[wave];
    unsigned short* G = lds_g[wave];
    unsigned short* S = lds_s[wave];
    const int rowbase = blockIdx.x * 32 + wave * 16;

    // ---- Layer 0 (1 -> 256) + tanh chain, elementwise ----
    // h0 = x*W0 + b0 ; g_raw = W0 ; s_raw = 0
    // a = tanh(h0) ; g = t'*W0 ; s = -2*a*t'*W0^2   (t' = 1-a^2)
    for (int m = 0; m < 16; ++m) {
        float xv = x[rowbase + m];
        for (int n = lane; n < HDIM; n += 32) {
            float w0 = W0[n];
            float h  = fmaf(xv, w0, b0[n]);
            float a  = fast_tanh(h);
            float t  = 1.0f - a * a;
            A[m * HDIM + n] = f2bf(a);
            G[m * HDIM + n] = f2bf(t * w0);
            S[m * HDIM + n] = f2bf(-2.0f * a * t * w0 * w0);
        }
    }
    asm volatile("s_wait_dscnt 0" ::: "memory");

    // ---- Hidden layers 1,2 (256 -> 256, WMMA) ----
    for (int lay = 0; lay < 2; ++lay) {
        const float* bl = (lay == 0) ? b1 : b2;
        const int wbase = lay * (8 * 16 * 32 * 16);

        // Pass 1: value path. Load all A-frags, then overwrite A in place with tanh(hW+b).
        v16bf Ah[8];
#pragma unroll
        for (int kt = 0; kt < 8; ++kt) Ah[kt] = load_afrag(A, l, hi, kt);
        asm volatile("s_wait_dscnt 0" ::: "memory");
        for (int nt = 0; nt < 16; ++nt) {
            float bias = bl[nt * 16 + l];
            v8f c;
#pragma unroll
            for (int r = 0; r < 8; ++r) c[r] = bias;
#pragma unroll
            for (int kt = 0; kt < 8; ++kt) {
                v16bf B = load_bfrag(packW, wbase, kt, nt, lane);
                c = __builtin_amdgcn_wmma_f32_16x16x32_bf16(false, Ah[kt], false, B,
                                                            (short)0, c, false, false);
            }
            // C layout: VGPR r -> (M = r+8*hi, N = nt*16 + l)
#pragma unroll
            for (int r = 0; r < 8; ++r) {
                float a = fast_tanh(c[r]);
                A[(r + 8 * hi) * HDIM + nt * 16 + l] = f2bf(a);
            }
        }
        asm volatile("s_wait_dscnt 0" ::: "memory");

        // Pass 2+3: first/second derivative paths, interleaved so raw g GEMM
        // result (cg) is live for the s update. Overwrite G,S in place.
        v16bf Ag[8], As[8];
#pragma unroll
        for (int kt = 0; kt < 8; ++kt) {
            Ag[kt] = load_afrag(G, l, hi, kt);
            As[kt] = load_afrag(S, l, hi, kt);
        }
        asm volatile("s_wait_dscnt 0" ::: "memory");
        for (int nt = 0; nt < 16; ++nt) {
            v8f cg, cs;
#pragma unroll
            for (int r = 0; r < 8; ++r) { cg[r] = 0.0f; cs[r] = 0.0f; }
#pragma unroll
            for (int kt = 0; kt < 8; ++kt) {
                v16bf B = load_bfrag(packW, wbase, kt, nt, lane);
                cg = __builtin_amdgcn_wmma_f32_16x16x32_bf16(false, Ag[kt], false, B,
                                                             (short)0, cg, false, false);
                cs = __builtin_amdgcn_wmma_f32_16x16x32_bf16(false, As[kt], false, B,
                                                             (short)0, cs, false, false);
            }
#pragma unroll
            for (int r = 0; r < 8; ++r) {
                int idx = (r + 8 * hi) * HDIM + nt * 16 + l;
                float a  = bf2f(A[idx]);         // new activation (pass 1 result)
                float t  = 1.0f - a * a;
                float gn = t * cg[r];
                float sn = t * cs[r] - 2.0f * a * cg[r] * gn;
                G[idx] = f2bf(gn);
                S[idx] = f2bf(sn);
            }
        }
        asm volatile("s_wait_dscnt 0" ::: "memory");
    }

    // ---- Final layer (256 -> 1): u and u_xx per point ----
    if (hi == 0) {
        const int m = l;
        float au = 0.0f, auxx = 0.0f;
        for (int n = 0; n < HDIM; ++n) {
            float w3 = W3[n];
            au   = fmaf(bf2f(A[m * HDIM + n]), w3, au);
            auxx = fmaf(bf2f(S[m * HDIM + n]), w3, auxx);
        }
        const int g = rowbase + m;
        float u  = au + b3[0];
        float xv = x[g];
        vout[g] = wq[g] * (auxx + __sinf(PI_F * xv));
        if (g == 0 || g == NPTS - 1)
            atomicAdd(&out[1], 5.0f * u * u);   // 0.5 * 10 * u^2 per endpoint
    }
}

// Partial r_k over an n-slice: partial[s][k] = sum_{n in slice s} sin(pi*(k+1)*x_n)*v_n.
// 64 slices x 1024 k = 65536 threads (2048 waves) instead of 32 waves.
__global__ __launch_bounds__(256) void vpinn_proj_kernel(
    const float* __restrict__ x, const float* __restrict__ v,
    float* __restrict__ partial)
{
    const int kb = blockIdx.x & 3;              // 4 blocks cover k
    const int s  = blockIdx.x >> 2;             // slice 0..NSPLIT-1
    const int k  = kb * 256 + threadIdx.x;      // 0..1023  (test index k+1)
    const float pk = PI_F * (float)(k + 1);
    const int n0 = s * (NPTS / NSPLIT);
    const int n1 = n0 + (NPTS / NSPLIT);
    float acc = 0.0f;
    for (int n = n0; n < n1; n += 4) {
        acc = fmaf(__sinf(pk * x[n + 0]), v[n + 0], acc);
        acc = fmaf(__sinf(pk * x[n + 1]), v[n + 1], acc);
        acc = fmaf(__sinf(pk * x[n + 2]), v[n + 2], acc);
        acc = fmaf(__sinf(pk * x[n + 3]), v[n + 3], acc);
    }
    partial[s * KTEST + k] = acc;
}

// r_k = sum_s partial[s][k]; loss_interior = mean_k r_k^2.
__global__ __launch_bounds__(256) void vpinn_reduce_kernel(
    const float* __restrict__ partial, float* __restrict__ out)
{
    const int k = blockIdx.x * 256 + threadIdx.x;   // 4 blocks
    float r = 0.0f;
#pragma unroll 4
    for (int s = 0; s < NSPLIT; ++s) r += partial[s * KTEST + k];
    float p = r * r * (1.0f / (float)KTEST);
#pragma unroll
    for (int off = 16; off > 0; off >>= 1) p += __shfl_down(p, off);
    if ((threadIdx.x & 31) == 0) atomicAdd(&out[0], p);
}

extern "C" void kernel_launch(void* const* d_in, const int* in_sizes, int n_in,
                              void* d_out, int out_size, void* d_ws, size_t ws_size,
                              hipStream_t stream) {
    (void)in_sizes; (void)n_in; (void)out_size; (void)ws_size;
    const float* x  = (const float*)d_in[0];
    const float* wq = (const float*)d_in[1];
    const float* W0 = (const float*)d_in[2];
    const float* b0 = (const float*)d_in[3];
    const float* W1 = (const float*)d_in[4];
    const float* b1 = (const float*)d_in[5];
    const float* W2 = (const float*)d_in[6];
    const float* b2 = (const float*)d_in[7];
    const float* W3 = (const float*)d_in[8];
    const float* b3 = (const float*)d_in[9];
    float* out = (float*)d_out;

    // Workspace: [0,256KB) packed bf16 weights; [256KB,384KB) v; [384KB,640KB) partials.
    unsigned short* packW = (unsigned short*)d_ws;
    float* v       = (float*)((char*)d_ws + (size_t)2 * 65536 * sizeof(unsigned short));
    float* partial = v + NPTS;

    zero_out_kernel<<<1, 32, 0, stream>>>(out);
    pack_weights_kernel<<<512, 256, 0, stream>>>(W1, W2, packW);
    vpinn_mlp_kernel<<<NPTS / 32, 64, 0, stream>>>(x, wq, W0, b0, b1, b2, W3, b3, packW, v, out);
    vpinn_proj_kernel<<<NSPLIT * 4, 256, 0, stream>>>(x, v, partial);
    vpinn_reduce_kernel<<<KTEST / 256, 256, 0, stream>>>(partial, out);
}
